// WanTemporalAttention_85564338471358
// MI455X (gfx1250) — matrix operand order repaired and checked by reading, not compiled, not run on hardware
//
#include <hip/hip_runtime.h>
#include <hip/hip_bf16.h>

// ---------------------------------------------------------------------------
// Types / helpers
// ---------------------------------------------------------------------------
typedef __attribute__((ext_vector_type(16))) _Float16 v16h;
typedef __attribute__((ext_vector_type(8)))  float    v8f;

union FragH {
    v16h v;
    unsigned int u[8];
};

__device__ inline v8f v8f_zero() {
    v8f z;
#pragma unroll
    for (int i = 0; i < 8; ++i) z[i] = 0.f;
    return z;
}

// Async global -> LDS copy of 16 bytes per lane (CDNA5 ASYNCcnt path).
__device__ inline void async_copy_b128(unsigned lds_off, const void* gptr) {
    asm volatile("global_load_async_to_lds_b128 %0, %1, off"
                 :: "v"(lds_off), "v"(gptr)
                 : "memory");
}

#define DIM_C    1024
#define N_HEADS  16
#define HEAD_D   64
#define WIN      256
#define SEQ_T    2048
#define BATCH    2
#define C3       (3 * DIM_C)

// ---------------------------------------------------------------------------
// Kernel 1: f32 -> f16 conversion (weights)
// ---------------------------------------------------------------------------
__global__ void cvt_f32_f16_kernel(const float* __restrict__ in,
                                   _Float16* __restrict__ out, size_t n) {
    size_t i = (size_t)blockIdx.x * 256u + threadIdx.x;
    if (i < n) out[i] = (_Float16)in[i];
}

// ---------------------------------------------------------------------------
// Kernel 2: RMSNorm over C + transpose [B,C,T] -> [B,T,C] f16
// ---------------------------------------------------------------------------
__global__ __launch_bounds__(256)
void rmsnorm_kernel(const float* __restrict__ x, const float* __restrict__ w,
                    _Float16* __restrict__ xn) {
    __shared__ float tile[8][DIM_C];   // 32 KB
    __shared__ float red[256];
    __shared__ float rstd[8];

    const int b   = blockIdx.y;
    const int t0  = blockIdx.x * 8;
    const int tid = threadIdx.x;

    // load phase: coalesced along t
    const int ts = tid & 7;
    const int cl = tid >> 3;
    float ss = 0.f;
    for (int c = cl; c < DIM_C; c += 32) {
        float v = x[((size_t)b * DIM_C + c) * SEQ_T + t0 + ts];
        tile[ts][c] = v;
        ss += v * v;
    }
    red[tid] = ss;
    __syncthreads();
    if (tid < 8) {
        float s = 0.f;
        for (int i = 0; i < 32; ++i) s += red[i * 8 + tid];
        rstd[tid] = rsqrtf(s / (float)DIM_C + 1e-5f);
    }
    __syncthreads();

    // write phase: coalesced along c
    const int ts2 = tid >> 5;
    const int c0  = tid & 31;
    const float r = rstd[ts2];
    _Float16* dst = xn + ((size_t)b * SEQ_T + t0 + ts2) * DIM_C;
    for (int c = c0; c < DIM_C; c += 32)
        dst[c] = (_Float16)(tile[ts2][c] * r * w[c]);
}

// ---------------------------------------------------------------------------
// Kernel 3/6: WMMA GEMM, double-buffered async global->LDS staging,
// pipeline manually unrolled x2 so LDS buffer indices are literals.
// out[M,N] = A[M,K] @ Bt[N,K]^T + bias
// MODE 0: store f16 row-major out[m*N+n]            (QKV)
// MODE 1: store f32 transposed [B,N,T] + residual   (proj + identity)
// block = 256 threads (8 waves), tile 128x128, K-step 32
// ---------------------------------------------------------------------------
template <int MODE>
__global__ __launch_bounds__(256)
void gemm_wmma_kernel(const _Float16* __restrict__ A,
                      const _Float16* __restrict__ Bt,
                      const float* __restrict__ bias,
                      _Float16* __restrict__ outH,
                      float* __restrict__ outF,
                      const float* __restrict__ resid,
                      int M, int N, int K) {
    __shared__ _Float16 As[2][128][40];   // 20 KB
    __shared__ _Float16 Bs[2][128][40];   // 20 KB

    const int tid  = threadIdx.x;
    const int lane = tid & 31;
    const int wave = tid >> 5;
    const int wm   = wave & 3;       // 4 waves along M (32 rows each)
    const int wn   = wave >> 2;      // 2 waves along N (64 cols each)
    const int mBase = blockIdx.x * 128;
    const int nBase = blockIdx.y * 128;

    const int ln  = lane & 15;
    const int hiA = (lane < 16) ? 0 : 8;    // A-layout K offset per lane half
    const int hiB = (lane < 16) ? 0 : 16;   // unused: B fragments use hiA like A

    v8f acc[2][4];
#pragma unroll
    for (int mt = 0; mt < 2; ++mt)
#pragma unroll
        for (int nt = 0; nt < 4; ++nt) acc[mt][nt] = v8f_zero();

    // staging assignment: 128 rows, 2 threads per row, 16 halves each
    const int lrow = tid >> 1;
    const int lcol = (tid & 1) * 16;
    const _Float16* gA = A  + (size_t)(mBase + lrow) * K + lcol;
    const _Float16* gB = Bt + (size_t)(nBase + lrow) * K + lcol;
    unsigned ldsA[2], ldsB[2];
#pragma unroll
    for (int bf = 0; bf < 2; ++bf) {
        ldsA[bf] = (unsigned)(uintptr_t)&As[bf][lrow][lcol];
        ldsB[bf] = (unsigned)(uintptr_t)&Bs[bf][lrow][lcol];
    }

#define STAGE_TILE(bf, kk)                                                    \
    do {                                                                      \
        async_copy_b128(ldsA[bf],      gA + (kk));                            \
        async_copy_b128(ldsA[bf] + 16, gA + (kk) + 8);                        \
        async_copy_b128(ldsB[bf],      gB + (kk));                            \
        async_copy_b128(ldsB[bf] + 16, gB + (kk) + 8);                        \
    } while (0)

    auto compute = [&](int buf) {
        FragH a[2], b[4];
#pragma unroll
        for (int mt = 0; mt < 2; ++mt) {
            const int row = wm * 32 + mt * 16 + ln;
#pragma unroll
            for (int w = 0; w < 8; ++w) {
                const int k = ((w & 3) * 2) + ((w >> 2) * 16) + hiA;
                a[mt].u[w] =
                    *reinterpret_cast<const unsigned int*>(&As[buf][row][k]);
            }
        }
#pragma unroll
        for (int nt = 0; nt < 4; ++nt) {
            const int n = wn * 64 + nt * 16 + ln;
#pragma unroll
            for (int w = 0; w < 8; ++w) {
                const int k = ((w & 3) * 2) + ((w >> 2) * 16) + hiA;
                b[nt].u[w] =
                    *reinterpret_cast<const unsigned int*>(&Bs[buf][n][k]);
            }
        }
#pragma unroll
        for (int mt = 0; mt < 2; ++mt)
#pragma unroll
            for (int nt = 0; nt < 4; ++nt)
                acc[mt][nt] = __builtin_amdgcn_wmma_f32_16x16x32_f16(
                    false, a[mt].v, false, b[nt].v, (short)0, acc[mt][nt],
                    false, false);
    };

    STAGE_TILE(0, 0);
    // K/32 is even for all our shapes; process two k-steps per trip.
    for (int k0 = 0; k0 < K; k0 += 64) {
        // ---- half-step A: consume buffer 0, stage buffer 1 ----
        if (k0 + 32 < K) {
            STAGE_TILE(1, k0 + 32);
            asm volatile("s_wait_asynccnt 0x4" ::: "memory");
        } else {
            asm volatile("s_wait_asynccnt 0x0" ::: "memory");
        }
        __syncthreads();
        compute(0);
        __syncthreads();
        if (k0 + 32 >= K) break;

        // ---- half-step B: consume buffer 1, stage buffer 0 ----
        if (k0 + 64 < K) {
            STAGE_TILE(0, k0 + 64);
            asm volatile("s_wait_asynccnt 0x4" ::: "memory");
        } else {
            asm volatile("s_wait_asynccnt 0x0" ::: "memory");
        }
        __syncthreads();
        compute(1);
        __syncthreads();
    }
#undef STAGE_TILE

    // epilogue
#pragma unroll
    for (int mt = 0; mt < 2; ++mt) {
#pragma unroll
        for (int nt = 0; nt < 4; ++nt) {
            const int n  = nBase + wn * 64 + nt * 16 + ln;
            const float bv = bias[n];
            if (MODE == 0) {
#pragma unroll
                for (int r = 0; r < 8; ++r) {
                    const int m = mBase + wm * 32 + mt * 16 + r + hiA;
                    outH[(size_t)m * N + n] = (_Float16)(acc[mt][nt][r] + bv);
                }
            } else {
                // 8 consecutive t per lane -> two aligned float4 transactions
                const int m0 = mBase + wm * 32 + mt * 16 + hiA;
                const int bb = m0 / SEQ_T;
                const int t0 = m0 - bb * SEQ_T;
                const size_t oi = ((size_t)bb * N + n) * SEQ_T + t0;
                const float4* rsrc =
                    reinterpret_cast<const float4*>(resid + oi);
                float4* dst = reinterpret_cast<float4*>(outF + oi);
                float4 r0 = rsrc[0], r1 = rsrc[1];
                float4 o0, o1;
                o0.x = acc[mt][nt][0] + bv + r0.x;
                o0.y = acc[mt][nt][1] + bv + r0.y;
                o0.z = acc[mt][nt][2] + bv + r0.z;
                o0.w = acc[mt][nt][3] + bv + r0.w;
                o1.x = acc[mt][nt][4] + bv + r1.x;
                o1.y = acc[mt][nt][5] + bv + r1.y;
                o1.z = acc[mt][nt][6] + bv + r1.z;
                o1.w = acc[mt][nt][7] + bv + r1.w;
                dst[0] = o0;
                dst[1] = o1;
            }
        }
    }
}

// ---------------------------------------------------------------------------
// Kernel 4: gather V transposed: vt[b][h][d][t] = qkv[(b*T+t)*3C + 2C + h*64 + d]
// ---------------------------------------------------------------------------
__global__ void extract_vt_kernel(const _Float16* __restrict__ qkv,
                                  _Float16* __restrict__ vt) {
    size_t i = (size_t)blockIdx.x * 256u + threadIdx.x;
    const size_t total = (size_t)BATCH * N_HEADS * HEAD_D * SEQ_T;
    if (i >= total) return;
    const int t = (int)(i % SEQ_T);
    size_t r = i / SEQ_T;
    const int d = (int)(r % HEAD_D); r /= HEAD_D;
    const int h = (int)(r % N_HEADS);
    const int b = (int)(r / N_HEADS);
    vt[i] = qkv[((size_t)b * SEQ_T + t) * C3 + 2 * DIM_C + h * HEAD_D + d];
}

// ---------------------------------------------------------------------------
// Kernel 5: windowed-causal flash attention with WMMA
// block = 256 threads = 8 waves; each wave owns 16 query rows of one (b,h)
// ---------------------------------------------------------------------------
__global__ __launch_bounds__(256)
void attn_wmma_kernel(const _Float16* __restrict__ qkv,
                      const _Float16* __restrict__ vt,
                      _Float16* __restrict__ outH) {
    __shared__ float    sS[8][16][32];   // 16 KB score scratch
    __shared__ _Float16 sP[8][16][32];   //  8 KB prob scratch
    __shared__ float    sAl[8][16];      // per-row broadcast

    const int tid  = threadIdx.x;
    const int lane = tid & 31;
    const int wave = tid >> 5;
    const int bh   = blockIdx.y;
    const int b    = bh / N_HEADS;
    const int h    = bh % N_HEADS;
    const int q0   = blockIdx.x * 128 + wave * 16;

    const int ln  = lane & 15;
    const int hiA = (lane < 16) ? 0 : 8;
    const int hiB = (lane < 16) ? 0 : 16;
    const float NEG_INF = -__builtin_inff();

    // Q fragments (16 rows x 64 d, two K-steps of 32)
    FragH aQ[2];
    {
        const _Float16* qrow =
            qkv + ((size_t)b * SEQ_T + (q0 + ln)) * C3 + h * HEAD_D;
#pragma unroll
        for (int kk = 0; kk < 2; ++kk)
#pragma unroll
            for (int w = 0; w < 8; ++w) {
                const int d = kk * 32 + ((w & 3) * 2) + ((w >> 2) * 16) + hiA;
                aQ[kk].u[w] = *reinterpret_cast<const unsigned int*>(&qrow[d]);
            }
    }

    v8f accO[4];
#pragma unroll
    for (int nt = 0; nt < 4; ++nt) accO[nt] = v8f_zero();
    float mrun = NEG_INF, lrun = 0.f;
    const int rowq = q0 + ln;

    int kLo = q0 - (WIN - 1);
    if (kLo < 0) kLo = 0;
    kLo &= ~31;
    const int kHi = q0 + 15;

    for (int kc = kLo; kc <= kHi; kc += 32) {
        // prefetch next chunk's K and V cachelines (near temporal)
        if (kc + 32 <= kHi) {
            const _Float16* nk =
                qkv + ((size_t)b * SEQ_T + (kc + 32 + ln)) * C3 +
                DIM_C + h * HEAD_D;
            __builtin_prefetch(nk, 0, 3);
            const _Float16* nv =
                vt + (((size_t)b * N_HEADS + h) * HEAD_D + ln) * SEQ_T +
                (kc + 32);
            __builtin_prefetch(nv, 0, 3);
        }

        // ---- load all K fragments up front, then issue the WMMA group ----
        FragH bK[2][2];   // [key-tile][d-step]
#pragma unroll
        for (int kt = 0; kt < 2; ++kt) {
            const _Float16* krow =
                qkv + ((size_t)b * SEQ_T + (kc + kt * 16 + ln)) * C3 +
                DIM_C + h * HEAD_D;
#pragma unroll
            for (int dd = 0; dd < 2; ++dd)
#pragma unroll
                for (int w = 0; w < 8; ++w) {
                    const int d = dd * 32 + ((w & 3) * 2) + ((w >> 2) * 16) + hiA;
                    bK[kt][dd].u[w] =
                        *reinterpret_cast<const unsigned int*>(&krow[d]);
                }
        }
        v8f sAcc[2];
#pragma unroll
        for (int kt = 0; kt < 2; ++kt) {
            sAcc[kt] = v8f_zero();
            sAcc[kt] = __builtin_amdgcn_wmma_f32_16x16x32_f16(
                false, aQ[0].v, false, bK[kt][0].v, (short)0, sAcc[kt],
                false, false);
            sAcc[kt] = __builtin_amdgcn_wmma_f32_16x16x32_f16(
                false, aQ[1].v, false, bK[kt][1].v, (short)0, sAcc[kt],
                false, false);
        }
        // spill C-layout scores to wave-private LDS (scaled)
#pragma unroll
        for (int kt = 0; kt < 2; ++kt)
#pragma unroll
            for (int r = 0; r < 8; ++r)
                sS[wave][r + hiA][kt * 16 + ln] = sAcc[kt][r] * 0.125f;
        asm volatile("s_wait_dscnt 0" ::: "memory");

        // ---- masked online softmax: lane handles row=ln, half the cols ----
        const int colh = lane >> 4;
        float sv[16];
        float cmax = NEG_INF;
#pragma unroll
        for (int c = 0; c < 16; ++c) {
            const int col = colh * 16 + c;
            const int j = kc + col;
            float s = sS[wave][ln][col];
            const bool valid = (j <= rowq) && (j >= rowq - (WIN - 1));
            s = valid ? s : NEG_INF;
            sv[c] = s;
            cmax = fmaxf(cmax, s);
        }
        cmax = fmaxf(cmax, __shfl_xor(cmax, 16, 32));
        const float mnew = fmaxf(mrun, cmax);
        const float alpha = (mrun == NEG_INF) ? 0.f : __expf(mrun - mnew);

        float psum = 0.f;
#pragma unroll
        for (int c = 0; c < 16; ++c) {
            const float p = (sv[c] == NEG_INF || mnew == NEG_INF)
                                ? 0.f
                                : __expf(sv[c] - mnew);
            psum += p;
            sP[wave][ln][colh * 16 + c] = (_Float16)p;
        }
        psum += __shfl_xor(psum, 16, 32);
        lrun = lrun * alpha + psum;
        mrun = mnew;
        if (lane < 16) sAl[wave][ln] = alpha;
        asm volatile("s_wait_dscnt 0" ::: "memory");

        // rescale O accumulators by per-row alpha
#pragma unroll
        for (int r = 0; r < 8; ++r) {
            const float al = sAl[wave][r + hiA];
#pragma unroll
            for (int nt = 0; nt < 4; ++nt) accO[nt][r] *= al;
        }

        // ---- O += P V: load all V fragments, then the WMMA group ----
        FragH aP;
#pragma unroll
        for (int w = 0; w < 8; ++w) {
            const int k = ((w & 3) * 2) + ((w >> 2) * 16) + hiA;
            aP.u[w] = *reinterpret_cast<const unsigned int*>(&sP[wave][ln][k]);
        }
        FragH bV[4];
#pragma unroll
        for (int nt = 0; nt < 4; ++nt) {
            const int d = nt * 16 + ln;
            const _Float16* vrow =
                vt + (((size_t)b * N_HEADS + h) * HEAD_D + d) * SEQ_T;
#pragma unroll
            for (int w = 0; w < 8; ++w) {
                const int kk = kc + ((w & 3) * 2) + ((w >> 2) * 16) + hiA;
                bV[nt].u[w] = *reinterpret_cast<const unsigned int*>(&vrow[kk]);
            }
        }
#pragma unroll
        for (int nt = 0; nt < 4; ++nt)
            accO[nt] = __builtin_amdgcn_wmma_f32_16x16x32_f16(
                false, aP.v, false, bV[nt].v, (short)0, accO[nt], false, false);
    }

    // ---- finalize: O /= l, store f16 [B*T][C] ----
    if (lane < 16) sAl[wave][ln] = 1.f / lrun;
    asm volatile("s_wait_dscnt 0" ::: "memory");
#pragma unroll
    for (int nt = 0; nt < 4; ++nt) {
        const int d = nt * 16 + ln;
#pragma unroll
        for (int r = 0; r < 8; ++r) {
            const float inv = sAl[wave][r + hiA];
            const int q = q0 + r + hiA;
            outH[((size_t)b * SEQ_T + q) * DIM_C + h * HEAD_D + d] =
                (_Float16)(accO[nt][r] * inv);
        }
    }
}

// ---------------------------------------------------------------------------
// Launcher
// ---------------------------------------------------------------------------
extern "C" void kernel_launch(void* const* d_in, const int* in_sizes, int n_in,
                              void* d_out, int out_size, void* d_ws,
                              size_t ws_size, hipStream_t stream) {
    const float* x      = (const float*)d_in[0];  // [B,C,T]
    const float* norm_w = (const float*)d_in[1];  // [C]
    const float* qkv_w  = (const float*)d_in[2];  // [3C,C]
    const float* qkv_b  = (const float*)d_in[3];  // [3C]
    const float* proj_w = (const float*)d_in[4];  // [C,C]
    const float* proj_b = (const float*)d_in[5];  // [C]
    float* out = (float*)d_out;                   // [B,C,T]

    char* ws = (char*)d_ws;
    _Float16* qkv_w_h  = (_Float16*)ws;  ws += (size_t)C3 * DIM_C * 2;
    _Float16* proj_w_h = (_Float16*)ws;  ws += (size_t)DIM_C * DIM_C * 2;
    _Float16* xn_h     = (_Float16*)ws;  ws += (size_t)BATCH * SEQ_T * DIM_C * 2;
    _Float16* qkv_h    = (_Float16*)ws;  ws += (size_t)BATCH * SEQ_T * C3 * 2;
    _Float16* vt_h     = (_Float16*)ws;  ws += (size_t)BATCH * N_HEADS * HEAD_D * SEQ_T * 2;
    _Float16* attn_h   = (_Float16*)ws;  ws += (size_t)BATCH * SEQ_T * DIM_C * 2;

    const int M = BATCH * SEQ_T;  // 4096

    // 1. weight conversion
    {
        size_t n1 = (size_t)C3 * DIM_C;
        cvt_f32_f16_kernel<<<(unsigned)((n1 + 255) / 256), 256, 0, stream>>>(
            qkv_w, qkv_w_h, n1);
        size_t n2 = (size_t)DIM_C * DIM_C;
        cvt_f32_f16_kernel<<<(unsigned)((n2 + 255) / 256), 256, 0, stream>>>(
            proj_w, proj_w_h, n2);
    }
    // 2. RMSNorm + transpose
    rmsnorm_kernel<<<dim3(SEQ_T / 8, BATCH), 256, 0, stream>>>(x, norm_w, xn_h);
    // 3. QKV GEMM (M x 3C x C)
    gemm_wmma_kernel<0><<<dim3(M / 128, C3 / 128), 256, 0, stream>>>(
        xn_h, qkv_w_h, qkv_b, qkv_h, nullptr, nullptr, M, C3, DIM_C);
    // 4. V transpose gather
    {
        size_t nv = (size_t)BATCH * N_HEADS * HEAD_D * SEQ_T;
        extract_vt_kernel<<<(unsigned)((nv + 255) / 256), 256, 0, stream>>>(
            qkv_h, vt_h);
    }
    // 5. attention
    attn_wmma_kernel<<<dim3(SEQ_T / 128, BATCH * N_HEADS), 256, 0, stream>>>(
        qkv_h, vt_h, attn_h);
    // 6. output projection + residual, transposed store
    gemm_wmma_kernel<1><<<dim3(M / 128, DIM_C / 128), 256, 0, stream>>>(
        attn_h, proj_w_h, proj_b, nullptr, out, x, M, DIM_C, DIM_C);
}
